// FlashLTC_24043226923371
// MI455X (gfx1250) — matrix operand order repaired and checked by reading, not compiled, not run on hardware
//
#include <hip/hip_runtime.h>
#include <hip/hip_bf16.h>

// ---------------------------------------------------------------------------
// FlashLTC for MI455X (gfx1250): bf16 WMMA GEMMs, LDS-resident ODE scan.
//   B=8192, IN_DIM=HID=1024, ODE_STEPS=6, DT=1.0
//
// v2 changes (L2-traffic roofline):
//  - M-tile per block raised 16 -> 32: halves per-step weight re-reads
//    (256 blocks x 6 steps x 4MB = 6.1 GB from L2 instead of 12.3 GB),
//    and each B fetch now feeds 2 M-subtile WMMAs (4 WMMAs / K-chunk).
//  - f32 h state is thread-private RMW -> single LDS buffer; only the bf16
//    A-operand mirror is double-buffered. 263 KB LDS < 320 KB WGP budget.
// ---------------------------------------------------------------------------

#define B_SZ      8192
#define IN_DIM    1024
#define HID       1024
#define ODE_STEPS 6
#define MTILE     32          // batch rows per block (2 WMMA M-subtiles)

typedef __attribute__((ext_vector_type(16))) __bf16 v16bf;
typedef __attribute__((ext_vector_type(8)))  __bf16 v8bf;
typedef __attribute__((ext_vector_type(8)))  float  v8f;

// ---- workspace layout (bytes) ---------------------------------------------
#define OFF_XBF    ((size_t)0)                         // B*IN_DIM  bf16  16 MB
#define OFF_WIN    (OFF_XBF  + (size_t)B_SZ*IN_DIM*2)  // HID*IN_DIM bf16  2 MB
#define OFF_WGX    (OFF_WIN  + (size_t)HID*IN_DIM*2)
#define OFF_WGH    (OFF_WGX  + (size_t)HID*IN_DIM*2)
#define OFF_WREC   (OFF_WGH  + (size_t)HID*HID*2)
#define OFF_DRIVE  (OFF_WREC + (size_t)HID*HID*2)      // B*HID f32  32 MB
#define OFF_GATEX  (OFF_DRIVE + (size_t)B_SZ*HID*4)    // B*HID f32  32 MB
#define OFF_SCALE  (OFF_GATEX + (size_t)B_SZ*HID*4)    // HID f32

__device__ __forceinline__ unsigned short f2bf(float f) {
  unsigned int u = __float_as_uint(f);
  unsigned int r = (u + 0x7FFFu + ((u >> 16) & 1u)) >> 16;
  return (unsigned short)r;
}

__device__ __forceinline__ float fast_sigmoid(float x) {
  return 1.0f / (1.0f + __expf(-x));
}

// ---------------------------------------------------------------------------
__global__ void cvt_f32_bf16(const float* __restrict__ in,
                             unsigned short* __restrict__ out, int n) {
  int i = blockIdx.x * blockDim.x + threadIdx.x;
  int stride = gridDim.x * blockDim.x;
  for (; i < n; i += stride) out[i] = f2bf(in[i]);
}

__global__ void split_gate(const float* __restrict__ wg,   // HID x (IN_DIM+HID)
                           unsigned short* __restrict__ wgx,
                           unsigned short* __restrict__ wgh) {
  const int W = IN_DIM + HID;
  int i = blockIdx.x * blockDim.x + threadIdx.x;
  int stride = gridDim.x * blockDim.x;
  for (; i < HID * W; i += stride) {
    int r = i / W, c = i - r * W;
    unsigned short v = f2bf(wg[i]);
    if (c < IN_DIM) wgx[r * IN_DIM + c] = v;
    else            wgh[r * HID + (c - IN_DIM)] = v;
  }
}

__global__ void make_scale(const float* __restrict__ log_tau,
                           float* __restrict__ scale) {
  int i = blockIdx.x * blockDim.x + threadIdx.x;
  if (i < HID) scale[i] = (1.0f / (float)ODE_STEPS) * __expf(-log_tau[i]);
}

// ---------------------------------------------------------------------------
// A-operand (16x32 bf16): lanes 0-15 M=lane (K=k0+0..15),
// lanes 16-31 M=lane-16 (K=k0+16..31) -> two 16B loads per lane.
__device__ __forceinline__ v16bf load_a(const unsigned short* rowp, int k0,
                                        int half) {
  v8bf lo = *(const v8bf*)(rowp + k0 + half * 16);
  v8bf hi = *(const v8bf*)(rowp + k0 + half * 16 + 8);
  return __builtin_shufflevector(lo, hi, 0, 1, 2, 3, 4, 5, 6, 7,
                                         8, 9, 10, 11, 12, 13, 14, 15);
}

// B-operand (32x16 bf16): lane = N column; lanes 0-15 K=k0..k0+15,
// lanes 16-31 K=k0+16..k0+31 -> one contiguous 32B load per lane.
__device__ __forceinline__ v16bf load_b(const unsigned short* rowp, int k0) {
  return *(const v16bf*)(rowp + k0);
}

#define WMMA_BF16(a, b, c)                                                    \
  __builtin_amdgcn_wmma_f32_16x16x32_bf16(false, (a), false, (b), (short)0,   \
                                          (c), false, false)

// ---------------------------------------------------------------------------
// Phase 1: drive = x @ W_in.T + b_in ; gate_x = x @ Wg_x.T + b_g
// grid = B/32 blocks of 256 threads; wave -> 8 N-tiles x 2 M-subtiles.
__global__ __launch_bounds__(256) void ltc_in_gemm(
    const unsigned short* __restrict__ xbf,
    const unsigned short* __restrict__ win,
    const unsigned short* __restrict__ wgx,
    const float* __restrict__ b_in,
    const float* __restrict__ b_g,
    float* __restrict__ drive,
    float* __restrict__ gatex) {
  const int m0   = blockIdx.x * MTILE;
  const int lane = threadIdx.x & 31;
  const int wave = threadIdx.x >> 5;
  const int half = lane >> 4;
  const int l16  = lane & 15;

  const unsigned short* arow0 = xbf + (size_t)(m0 + l16) * IN_DIM;
  const unsigned short* arow1 = xbf + (size_t)(m0 + 16 + l16) * IN_DIM;

  for (int t = 0; t < 8; ++t) {
    const int n0 = (wave * 8 + t) * 16;
    const int n  = n0 + l16;
    const unsigned short* b0row = win + (size_t)n * IN_DIM + half * 16;
    const unsigned short* b1row = wgx + (size_t)n * IN_DIM + half * 16;
    const float bd = b_in[n], bg = b_g[n];
    v8f cd0, cg0, cd1, cg1;
#pragma unroll
    for (int v = 0; v < 8; ++v) {
      cd0[v] = bd; cg0[v] = bg; cd1[v] = bd; cg1[v] = bg;
    }

#pragma unroll 4
    for (int k0 = 0; k0 < IN_DIM; k0 += 32) {
      v16bf bw = load_b(b0row, k0);          // shared across both M-subtiles
      v16bf bx = load_b(b1row, k0);
      v16bf a0 = load_a(arow0, k0, half);
      v16bf a1 = load_a(arow1, k0, half);
      cd0 = WMMA_BF16(a0, bw, cd0);
      cg0 = WMMA_BF16(a0, bx, cg0);
      cd1 = WMMA_BF16(a1, bw, cd1);
      cg1 = WMMA_BF16(a1, bx, cg1);
    }
    const int mrow = half * 8;
#pragma unroll
    for (int v = 0; v < 8; ++v) {
      size_t i0 = (size_t)(m0 + mrow + v) * HID + n;
      size_t i1 = (size_t)(m0 + 16 + mrow + v) * HID + n;
      drive[i0] = cd0[v];
      gatex[i0] = cg0[v];
      drive[i1] = cd1[v];
      gatex[i1] = cg1[v];
    }
  }
}

// ---------------------------------------------------------------------------
// Phase 2: per-32-row persistent ODE scan.
// LDS: single f32 h state (thread-private RMW) + double-buffered bf16 mirror.
#define HF_STRIDE  1025
#define HB_STRIDE  1032
#define SCAN_LDS_BYTES \
  (MTILE * HF_STRIDE * 4 + 2 * MTILE * HB_STRIDE * 2)   // 263,296 B

__global__ __launch_bounds__(256) void ltc_scan(
    const float* __restrict__ h_in,
    const unsigned short* __restrict__ wgh,
    const unsigned short* __restrict__ wrec,
    const float* __restrict__ drive,
    const float* __restrict__ gatex,
    const float* __restrict__ scale,
    float* __restrict__ out) {
  extern __shared__ char smem[];
  float*          hf  = (float*)smem;                         // 32 x 1025 f32
  unsigned short* hb0 = (unsigned short*)(hf + MTILE * HF_STRIDE);
  unsigned short* hb1 = hb0 + MTILE * HB_STRIDE;

  const int m0   = blockIdx.x * MTILE;
  const int lane = threadIdx.x & 31;
  const int wave = threadIdx.x >> 5;
  const int half = lane >> 4;
  const int l16  = lane & 15;

  // stage initial h tile into LDS (f32 + bf16 mirror)
  for (int i = threadIdx.x; i < MTILE * 1024; i += 256) {
    int r = i >> 10, c = i & 1023;
    float v = h_in[(size_t)(m0 + r) * HID + c];
    hf[r * HF_STRIDE + c] = v;
    hb0[r * HB_STRIDE + c] = f2bf(v);
  }
  __syncthreads();

  int cur = 0;
  for (int step = 0; step < ODE_STEPS; ++step) {
    const unsigned short* hbc = cur ? hb1 : hb0;
    unsigned short*       hbn = cur ? hb0 : hb1;
    const unsigned short* arow0 = hbc + l16 * HB_STRIDE;
    const unsigned short* arow1 = hbc + (16 + l16) * HB_STRIDE;

    for (int t = 0; t < 8; ++t) {
      const int n0 = (wave * 8 + t) * 16;
      const int n  = n0 + l16;
      const float sc = scale[n];
      const unsigned short* b0row = wgh  + (size_t)n * HID + half * 16;
      const unsigned short* b1row = wrec + (size_t)n * HID + half * 16;
      const int mrow = half * 8;

      v8f cg0, cd0, cg1, cd1;
#pragma unroll
      for (int v = 0; v < 8; ++v) {
        size_t i0 = (size_t)(m0 + mrow + v) * HID + n;
        size_t i1 = (size_t)(m0 + 16 + mrow + v) * HID + n;
        cg0[v] = gatex[i0];
        cd0[v] = drive[i0];
        cg1[v] = gatex[i1];
        cd1[v] = drive[i1];
      }

#pragma unroll 4
      for (int k0 = 0; k0 < HID; k0 += 32) {
        __builtin_prefetch(b0row + k0 + 512, 0, 3);   // global_prefetch_b8
        __builtin_prefetch(b1row + k0 + 512, 0, 3);
        v16bf bg = load_b(b0row, k0);     // one B fetch feeds 2 M-subtiles
        v16bf br = load_b(b1row, k0);
        v16bf a0 = load_a(arow0, k0, half);   // ds_load_b128 x2
        v16bf a1 = load_a(arow1, k0, half);
        cg0 = WMMA_BF16(a0, bg, cg0);
        cd0 = WMMA_BF16(a0, br, cd0);
        cg1 = WMMA_BF16(a1, bg, cg1);
        cd1 = WMMA_BF16(a1, br, cd1);
      }

      // pointwise LTC update: h' = h + scale * (sigmoid(g)*tanh(d) - h)
#pragma unroll
      for (int v = 0; v < 8; ++v) {
        int ma = mrow + v;
        int mb = 16 + mrow + v;
        float holda = hf[ma * HF_STRIDE + n];
        float holdb = hf[mb * HF_STRIDE + n];
        float ga = fast_sigmoid(cg0[v]);
        float gb = fast_sigmoid(cg1[v]);
        float aa = tanhf(cd0[v]);
        float ab = tanhf(cd1[v]);
        float hna = fmaf(sc, ga * aa - holda, holda);
        float hnb = fmaf(sc, gb * ab - holdb, holdb);
        hf[ma * HF_STRIDE + n] = hna;       // thread-private RMW: no race
        hf[mb * HF_STRIDE + n] = hnb;
        hbn[ma * HB_STRIDE + n] = f2bf(hna);
        hbn[mb * HB_STRIDE + n] = f2bf(hnb);
      }
    }
    __syncthreads();   // one barrier per step (bf16 mirror double-buffered)
    cur ^= 1;
  }

  // emit h to both tuple slots
  for (int i = threadIdx.x; i < MTILE * 1024; i += 256) {
    int r = i >> 10, c = i & 1023;
    float v = hf[r * HF_STRIDE + c];
    size_t o = (size_t)(m0 + r) * HID + c;
    out[o] = v;
    out[(size_t)B_SZ * HID + o] = v;
  }
}

// ---------------------------------------------------------------------------
extern "C" void kernel_launch(void* const* d_in, const int* in_sizes, int n_in,
                              void* d_out, int out_size, void* d_ws,
                              size_t ws_size, hipStream_t stream) {
  const float* x       = (const float*)d_in[0];
  const float* h       = (const float*)d_in[1];
  const float* log_tau = (const float*)d_in[2];
  const float* Win_w   = (const float*)d_in[3];
  const float* Win_b   = (const float*)d_in[4];
  const float* Wrec_w  = (const float*)d_in[5];
  const float* Wg_w    = (const float*)d_in[6];
  const float* Wg_b    = (const float*)d_in[7];
  float* out = (float*)d_out;

  char* ws = (char*)d_ws;
  unsigned short* xbf   = (unsigned short*)(ws + OFF_XBF);
  unsigned short* winb  = (unsigned short*)(ws + OFF_WIN);
  unsigned short* wgxb  = (unsigned short*)(ws + OFF_WGX);
  unsigned short* wghb  = (unsigned short*)(ws + OFF_WGH);
  unsigned short* wrecb = (unsigned short*)(ws + OFF_WREC);
  float* drive = (float*)(ws + OFF_DRIVE);
  float* gatex = (float*)(ws + OFF_GATEX);
  float* scl   = (float*)(ws + OFF_SCALE);

  cvt_f32_bf16<<<2048, 256, 0, stream>>>(x, xbf, B_SZ * IN_DIM);
  cvt_f32_bf16<<<512, 256, 0, stream>>>(Win_w, winb, HID * IN_DIM);
  cvt_f32_bf16<<<512, 256, 0, stream>>>(Wrec_w, wrecb, HID * HID);
  split_gate<<<1024, 256, 0, stream>>>(Wg_w, wgxb, wghb);
  make_scale<<<(HID + 255) / 256, 256, 0, stream>>>(log_tau, scl);

  ltc_in_gemm<<<B_SZ / MTILE, 256, 0, stream>>>(xbf, winb, wgxb, Win_b, Wg_b,
                                                drive, gatex);
  ltc_scan<<<B_SZ / MTILE, 256, SCAN_LDS_BYTES, stream>>>(h, wghb, wrecb,
                                                          drive, gatex, scl,
                                                          out);
}
